// MultiHeadAttention_83777632076280
// MI455X (gfx1250) — matrix operand order repaired and checked by reading, not compiled, not run on hardware
//
#include <hip/hip_runtime.h>
#include <hip/hip_bf16.h>

typedef __bf16 bf16_t;
typedef __attribute__((ext_vector_type(8)))  __bf16 v8bf;
typedef __attribute__((ext_vector_type(16))) __bf16 v16bf;
typedef __attribute__((ext_vector_type(8)))  float  v8f;

#define B_   4
#define S_   1024
#define D_   1024
#define H_   16
#define HALF_ 512
#define M_   4096   // B*S

__device__ __forceinline__ v16bf pack16(const bf16_t* p0, const bf16_t* p1) {
  v8bf a = *(const v8bf*)p0;
  v8bf b = *(const v8bf*)p1;
  v16bf r;
#pragma unroll
  for (int i = 0; i < 8; ++i) { r[i] = a[i]; r[8 + i] = b[i]; }
  return r;
}

__device__ __forceinline__ v8f wmma_bf16(v16bf a, v16bf b, v8f c) {
  return __builtin_amdgcn_wmma_f32_16x16x32_bf16(false, a, false, b, (short)0, c,
                                                 false, false);
}

__device__ __forceinline__ v8f vzero8() {
  v8f c = {0.f, 0.f, 0.f, 0.f, 0.f, 0.f, 0.f, 0.f};
  return c;
}

// ---------------------------------------------------------------- converts
// src: (B,S,1024) fp32; dst: (4096,512) bf16 taking cols [col0, col0+512)
__global__ void convert_half_k(const float* __restrict__ src, bf16_t* __restrict__ dst,
                               int col0) {
  int i = blockIdx.x * 256 + threadIdx.x;          // 0 .. 2M-1
  int row = i >> 9, col = i & 511;
  dst[i] = (bf16_t)src[(size_t)row * 1024 + col0 + col];
}

// w: (512,512) fp32 row-major; dst[n*512+k] = bf16(w[k*512+n]) (transposed)
__global__ void convert_wt_k(const float* __restrict__ w, bf16_t* __restrict__ dst) {
  int i = blockIdx.x * 256 + threadIdx.x;          // 0 .. 256K-1
  int nn = i >> 9, k = i & 511;
  dst[i] = (bf16_t)w[(size_t)k * 512 + nn];
}

// ---------------------------------------------------------------- projections
// C(4096x512) = X(4096x512 bf16) @ W ; Wt is transposed bf16 [n][k].
// mode 0: out[b][h][s][32]   (Q1/K1)
// mode 1: out[b][h][doff+d][1024] = C   (V transposed, doff = 0 or 32)
__global__ void proj_gemm(const bf16_t* __restrict__ X, const bf16_t* __restrict__ Wt,
                          bf16_t* __restrict__ out, int mode, int doff) {
  const int tid = threadIdx.x, wave = tid >> 5, lane = tid & 31;
  const int half = lane >> 4, n = lane & 15;
  const int gw = blockIdx.x * 4 + wave;            // 8192 wave-tiles
  const int mt = gw >> 5;                          // 0..255
  const int nt = gw & 31;                          // 0..31

  const bf16_t* xrow = X + (size_t)(mt * 16 + n) * 512;   // A row m = lane&15
  const bf16_t* wrow = Wt + (size_t)(nt * 16 + n) * 512;  // B column n

  v8f c = vzero8();
#pragma unroll
  for (int kk = 0; kk < 16; ++kk) {
    if (kk + 1 < 16) __builtin_prefetch(xrow + (kk + 1) * 32, 0, 0);
    v16bf A = pack16(xrow + kk * 32 + half * 8, xrow + kk * 32 + half * 8 + 16);
    v16bf Bf = pack16(wrow + kk * 32 + half * 8, wrow + kk * 32 + half * 8 + 16);
    c = wmma_bf16(A, Bf, c);
  }

#pragma unroll
  for (int r = 0; r < 8; ++r) {
    int row = mt * 16 + r + half * 8;
    int col = nt * 16 + n;
    int bb = row >> 10, s = row & 1023;
    int hh = col >> 5, d = col & 31;
    if (mode == 0)
      out[(((size_t)bb * 16 + hh) * 1024 + s) * 32 + d] = (bf16_t)c[r];
    else
      out[(((size_t)bb * 16 + hh) * 64 + (doff + d)) * 1024 + s] = (bf16_t)c[r];
  }
}

// ---------------------------------------------------------------- attention
// Q1,K1: [b][h][s][32] bf16 ; Vt: [b][h][64][1024] bf16
// attn_out: [b][h][q][k] fp32 ; qkvb: [b][s][h*64+d] bf16
#define WAVES_ATT 4
__global__ void attn_kernel(const bf16_t* __restrict__ Q1, const bf16_t* __restrict__ K1,
                            const bf16_t* __restrict__ Vt, float* __restrict__ attn_out,
                            bf16_t* __restrict__ qkvb) {
  const int tid = threadIdx.x, wave = tid >> 5, lane = tid & 31;
  const int half = lane >> 4, n = lane & 15;
  const int gw = blockIdx.x * WAVES_ATT + wave;    // 4096 waves total
  const int qt = gw & 63;
  const int h = (gw >> 6) & 15;
  const int b = gw >> 10;
  const int bh = b * 16 + h;
  const int q0 = qt << 4;
  const int ncc = qt + 1;                          // causal key chunks of 16
  const float scale = 0.17677669529663687f;        // 1/sqrt(32)

  __shared__ __align__(16) bf16_t pstage[WAVES_ATT][16][32];

  // Q fragment: row m = lane&15, K chunks {koff..koff+7} and {koff+16..koff+23}
  const bf16_t* qrow = Q1 + ((size_t)bh * 1024 + q0 + n) * 32;
  const int koff = half * 8;
  const v16bf Aq = pack16(qrow + koff, qrow + koff + 16);

  // ---- pass 1: online row max / sum (per-lane partials over this lane's column)
  float mx[8], sm[8];
#pragma unroll
  for (int r = 0; r < 8; ++r) { mx[r] = -3.0e38f; sm[r] = 0.f; }

  for (int kc = 0; kc < ncc; ++kc) {
    const bf16_t* krow = K1 + ((size_t)bh * 1024 + kc * 16 + n) * 32;
    v16bf Bk = pack16(krow + half * 8, krow + half * 8 + 16);
    v8f c = wmma_bf16(Aq, Bk, vzero8());
    const int k = kc * 16 + n;
#pragma unroll
    for (int r = 0; r < 8; ++r) {
      int q = q0 + r + half * 8;
      float s = (k <= q) ? c[r] * scale : -1.0e9f;
      float nm = fmaxf(mx[r], s);
      sm[r] = sm[r] * __expf(mx[r] - nm) + __expf(s - nm);
      mx[r] = nm;
    }
  }
  // reduce across the 16 lanes holding the same rows (lanes 0-15: rows 0-7, 16-31: 8-15)
#pragma unroll
  for (int r = 0; r < 8; ++r) {
#pragma unroll
    for (int off = 1; off < 16; off <<= 1) {
      float om = __shfl_xor(mx[r], off, 32);
      float os = __shfl_xor(sm[r], off, 32);
      float nm = fmaxf(mx[r], om);
      sm[r] = sm[r] * __expf(mx[r] - nm) + os * __expf(om - nm);
      mx[r] = nm;
    }
    sm[r] = 1.0f / sm[r];
  }

  // ---- pass 2: recompute scores, write softmax, accumulate P @ V
  v8f acc[4];
#pragma unroll
  for (int nt = 0; nt < 4; ++nt) acc[nt] = vzero8();

  const int ncc2 = (ncc + 1) >> 1;                 // key chunks of 32
  for (int kc2 = 0; kc2 < ncc2; ++kc2) {
#pragma unroll
    for (int sub = 0; sub < 2; ++sub) {
      int kc = kc2 * 2 + sub;
      if (kc < ncc) {
        const bf16_t* krow = K1 + ((size_t)bh * 1024 + kc * 16 + n) * 32;
        v16bf Bk = pack16(krow + half * 8, krow + half * 8 + 16);
        v8f c = wmma_bf16(Aq, Bk, vzero8());
        const int k = kc * 16 + n;
#pragma unroll
        for (int r = 0; r < 8; ++r) {
          int q = q0 + r + half * 8;
          float s = (k <= q) ? c[r] * scale : -1.0e9f;
          float p = __expf(s - mx[r]) * sm[r];
          attn_out[((size_t)bh * 1024 + q) * 1024 + k] = p;
          pstage[wave][r + half * 8][sub * 16 + n] = (bf16_t)p;
        }
      } else {
#pragma unroll
        for (int r = 0; r < 8; ++r)
          pstage[wave][r + half * 8][sub * 16 + n] = (bf16_t)0.f;
      }
    }
    // wave-internal LDS transpose C-layout -> A-layout
    asm volatile("s_wait_dscnt 0" ::: "memory");
    const bf16_t* prow = &pstage[wave][n][0];
    v16bf Ap = pack16(prow + koff, prow + koff + 16);
#pragma unroll
    for (int nt = 0; nt < 4; ++nt) {
      const bf16_t* vrow =
          Vt + ((size_t)bh * 64 + nt * 16 + n) * 1024 + kc2 * 32 + half * 8;
      v16bf Bv = pack16(vrow, vrow + 16);
      acc[nt] = wmma_bf16(Ap, Bv, acc[nt]);
    }
  }

  // qkv -> [b][s][h*64+d] bf16
#pragma unroll
  for (int nt = 0; nt < 4; ++nt) {
#pragma unroll
    for (int r = 0; r < 8; ++r) {
      int q = q0 + r + half * 8;
      qkvb[((size_t)b * 1024 + q) * 1024 + h * 64 + nt * 16 + n] = (bf16_t)acc[nt][r];
    }
  }
  // zero masked tail of softmax_attn (exp(-1e9) underflows to exactly 0)
  for (int kc = ncc; kc < 64; ++kc) {
    int k = kc * 16 + n;
#pragma unroll
    for (int r = 0; r < 8; ++r) {
      int q = q0 + r + half * 8;
      attn_out[((size_t)bh * 1024 + q) * 1024 + k] = 0.f;
    }
  }
}

// ---------------------------------------------------------------- FC + layernorm
// out_half = qkvb[:, fh*512:(fh+1)*512] @ Wfc + residual, then LN over 512.
__global__ void fc_ln(const bf16_t* __restrict__ qkvb, const bf16_t* __restrict__ Wt1,
                      const bf16_t* __restrict__ Wt2, const float* __restrict__ inQ,
                      const float* __restrict__ g, const float* __restrict__ be,
                      float* __restrict__ out) {
  const int fh = blockIdx.y;
  const bf16_t* Wt = fh ? Wt2 : Wt1;
  const int mt = blockIdx.x;                       // 0..255
  const int tid = threadIdx.x, wave = tid >> 5, lane = tid & 31;
  const int half = lane >> 4, n = lane & 15;

  __shared__ float t[16][512];
  __shared__ float mrow[16], rrow[16];

  const bf16_t* arow = qkvb + (size_t)(mt * 16 + n) * 1024 + fh * 512;
  v8f acc[4];
#pragma unroll
  for (int nt = 0; nt < 4; ++nt) acc[nt] = vzero8();

#pragma unroll
  for (int kk = 0; kk < 16; ++kk) {
    v16bf A = pack16(arow + kk * 32 + half * 8, arow + kk * 32 + half * 8 + 16);
#pragma unroll
    for (int nt = 0; nt < 4; ++nt) {
      const bf16_t* wrow = Wt + (size_t)(wave * 64 + nt * 16 + n) * 512;
      v16bf Bf = pack16(wrow + kk * 32 + half * 8, wrow + kk * 32 + half * 8 + 16);
      acc[nt] = wmma_bf16(A, Bf, acc[nt]);
    }
  }

#pragma unroll
  for (int nt = 0; nt < 4; ++nt) {
#pragma unroll
    for (int r = 0; r < 8; ++r) {
      int m = r + half * 8;
      int col = wave * 64 + nt * 16 + n;
      int row = mt * 16 + m, bb = row >> 10, s = row & 1023;
      float res = inQ[((size_t)bb * 1024 + s) * 1024 + fh * 512 + col];
      t[m][col] = acc[nt][r] + res;
    }
  }
  __syncthreads();

  {
    int m = tid >> 4;                              // row 0..15, 16 threads each
    int l16 = tid & 15;
    float su = 0.f, sq = 0.f;
    for (int i = 0; i < 32; ++i) {
      float v = t[m][l16 + i * 16];
      su += v; sq += v * v;
    }
#pragma unroll
    for (int off = 1; off < 16; off <<= 1) {
      su += __shfl_xor(su, off, 32);
      sq += __shfl_xor(sq, off, 32);
    }
    if (l16 == 0) {
      float mean = su * (1.f / 512.f);
      float var = sq * (1.f / 512.f) - mean * mean;
      mrow[m] = mean;
      rrow[m] = rsqrtf(var + 1e-5f);
    }
  }
  __syncthreads();

  for (int i = 0; i < 32; ++i) {
    int idx = tid + i * 256;
    int m = idx >> 9, col = idx & 511;
    int row = mt * 16 + m, bb = row >> 10, s = row & 1023;
    float v = (t[m][col] - mrow[m]) * rrow[m] * g[col] + be[col];
    out[((size_t)bb * 1024 + s) * 1024 + fh * 512 + col] = v;
  }
}

// ---------------------------------------------------------------- launcher
extern "C" void kernel_launch(void* const* d_in, const int* in_sizes, int n_in,
                              void* d_out, int out_size, void* d_ws, size_t ws_size,
                              hipStream_t stream) {
  (void)in_sizes; (void)n_in; (void)out_size;
  const float* inQ  = (const float*)d_in[0];
  const float* inK  = (const float*)d_in[1];
  const float* inV  = (const float*)d_in[2];
  // d_in[3] = attn_mask (causal, known statically; unused)
  const float* WQ1  = (const float*)d_in[4];
  const float* WK1  = (const float*)d_in[5];
  const float* WV1  = (const float*)d_in[6];
  // d_in[7] = WQ2, d_in[8] = WK2 : dead code in the reference
  const float* WV2  = (const float*)d_in[9];
  const float* Wfc1 = (const float*)d_in[10];
  const float* Wfc2 = (const float*)d_in[11];
  const float* lng  = (const float*)d_in[12];
  const float* lnb  = (const float*)d_in[13];

  float* out = (float*)d_out;
  float* attn_out = out + (size_t)B_ * S_ * D_;    // softmax_attn follows output

  char* ws = (char*)d_ws;
  size_t off = 0;
  auto alloc = [&](size_t bytes) -> void* {
    void* p = ws + off;
    off += (bytes + 255) & ~(size_t)255;
    return p;
  };
  const size_t XHB = (size_t)M_ * HALF_ * sizeof(bf16_t);   // 4 MB
  const size_t WTB = (size_t)HALF_ * HALF_ * sizeof(bf16_t); // 512 KB

  bf16_t* Xq1 = (bf16_t*)alloc(XHB);
  bf16_t* Xk1 = (bf16_t*)alloc(XHB);
  bf16_t* Xv1 = (bf16_t*)alloc(XHB);
  bf16_t* Xv2 = (bf16_t*)alloc(XHB);
  bf16_t* WQ1t = (bf16_t*)alloc(WTB);
  bf16_t* WK1t = (bf16_t*)alloc(WTB);
  bf16_t* WV1t = (bf16_t*)alloc(WTB);
  bf16_t* WV2t = (bf16_t*)alloc(WTB);
  bf16_t* Wfc1t = (bf16_t*)alloc(WTB);
  bf16_t* Wfc2t = (bf16_t*)alloc(WTB);
  bf16_t* Q1b = (bf16_t*)alloc(XHB);                         // [b][h][s][32]
  bf16_t* K1b = (bf16_t*)alloc(XHB);
  bf16_t* Vtb = (bf16_t*)alloc(2 * XHB);                     // [b][h][64][1024]
  bf16_t* qkvb = (bf16_t*)alloc((size_t)M_ * D_ * sizeof(bf16_t));
  if (off > ws_size) return;                                 // workspace too small

  convert_half_k<<<8192, 256, 0, stream>>>(inQ, Xq1, 0);
  convert_half_k<<<8192, 256, 0, stream>>>(inK, Xk1, 0);
  convert_half_k<<<8192, 256, 0, stream>>>(inV, Xv1, 0);
  convert_half_k<<<8192, 256, 0, stream>>>(inV, Xv2, HALF_);
  convert_wt_k<<<1024, 256, 0, stream>>>(WQ1, WQ1t);
  convert_wt_k<<<1024, 256, 0, stream>>>(WK1, WK1t);
  convert_wt_k<<<1024, 256, 0, stream>>>(WV1, WV1t);
  convert_wt_k<<<1024, 256, 0, stream>>>(WV2, WV2t);
  convert_wt_k<<<1024, 256, 0, stream>>>(Wfc1, Wfc1t);
  convert_wt_k<<<1024, 256, 0, stream>>>(Wfc2, Wfc2t);

  proj_gemm<<<2048, 128, 0, stream>>>(Xq1, WQ1t, Q1b, 0, 0);
  proj_gemm<<<2048, 128, 0, stream>>>(Xk1, WK1t, K1b, 0, 0);
  proj_gemm<<<2048, 128, 0, stream>>>(Xv1, WV1t, Vtb, 1, 0);
  proj_gemm<<<2048, 128, 0, stream>>>(Xv2, WV2t, Vtb, 1, 32);

  attn_kernel<<<1024, WAVES_ATT * 32, 0, stream>>>(Q1b, K1b, Vtb, attn_out, qkvb);

  fc_ln<<<dim3(256, 2), 256, 0, stream>>>(qkvb, Wfc1t, Wfc2t, inQ, lng, lnb, out);
}
